// GeometryTransformation_4561255268407
// MI455X (gfx1250) — matrix operand order repaired and checked by reading, not compiled, not run on hardware
//
#include <hip/hip_runtime.h>

typedef __attribute__((ext_vector_type(16))) _Float16 v16h;
typedef __attribute__((ext_vector_type(8)))  float    v8f;
typedef __attribute__((ext_vector_type(4)))  _Float16 v4h;
typedef __attribute__((ext_vector_type(4)))  float    v4f;

#define NN 128
#define CC 16
#define VOX (NN * NN * NN)   /* 2097152 voxels */
#define NCHUNK 14            /* ceil(27*16 / 32) K-chunks */
#define TZ 2
#define TY 8
#define TX 16
#define LXc (TX + 2)         /* 18 */
#define LYc (TY + 2)         /* 10 */
#define LZc (TZ + 2)         /* 4  */
#define LPTS (LZc * LYc * LXc)  /* 720 halo points */

// Compile-time LDS offset (in halves) for a given tap index.
// tap 27 (K padding) clamps to 26: the packed A is zero there, so the
// multiplied garbage contributes 0 while staying finite (no NaN).
constexpr int dtab(int tap) {
    int t  = tap > 26 ? 26 : tap;
    int dz = t / 9, r = t - 9 * dz, dy = r / 3, dx = r - 3 * dy;
    return ((dz * LYc + dy) * LXc + dx) * CC;
}

// ---------------------------------------------------------------------------
// Pack conv weights (O=16, I=16, 3,3,3 f32) into the CDNA5 WMMA A-matrix
// per-lane layout for V_WMMA_F32_16X16X32_F16 (16-bit A 16x32, contiguous k order, same as B):
//   lane m:    K0..15 in elems 0..15
//   lane m+16: K16..31 in elems 0..15
// K-global = tap*16 + ic, padded to 14 chunks of 32.
// ---------------------------------------------------------------------------
__global__ void pack_weights_kernel(const float* __restrict__ w,
                                    _Float16* __restrict__ dst) {
    int t = blockIdx.x * blockDim.x + threadIdx.x;
    if (t >= NCHUNK * 32 * 16) return;
    int e    = t & 15;
    int lane = (t >> 4) & 31;
    int ch   = t >> 9;
    int hi   = lane >> 4;
    int kl   = hi * 16 + e;
    int k    = ch * 32 + kl;
    int tap  = k >> 4;
    int ic   = k & 15;
    int oc   = lane & 15;
    float v  = 0.0f;
    if (tap < 27) {
        int dz = tap / 9, r = tap - 9 * dz, dy = r / 3, dx = r - 3 * dy;
        v = w[(((oc * CC + ic) * 3 + dz) * 3 + dy) * 3 + dx];
    }
    dst[(ch * 32 + lane) * 16 + e] = (_Float16)v;
}

// ---------------------------------------------------------------------------
// Backprojection with single angle 0 collapses exactly to
// vol[c,z,y,x] = proj[c,0,z,y]. Write channel-last f16 volume.
// ---------------------------------------------------------------------------
__global__ void backproject_kernel(const float* __restrict__ proj,
                                   _Float16* __restrict__ X) {
    int v = blockIdx.x * blockDim.x + threadIdx.x;  // voxel index
    if (v >= VOX) return;
    int z = v >> 14;
    int y = (v >> 7) & 127;
    v16h out;
    #pragma unroll
    for (int c = 0; c < CC; ++c)
        out[c] = (_Float16)proj[(c * NN + z) * NN + y];
    *(v16h*)&X[(size_t)v * CC] = out;
}

// ---------------------------------------------------------------------------
// Implicit-GEMM 3D conv via V_WMMA_F32_16X16X32_F16.
// Workgroup = 8 waves, tile 2z x 8y x 16x outputs. Each wave computes TWO
// 16oc x 16x output rows sharing one A operand (28 WMMA / wave). Halo is
// staged in LDS channel-last so each B lane is one aligned 32-byte read.
// LDS offsets per chunk are compile-time constants selected by the lane's
// upper-half bit (one v_cndmask instead of per-lane integer division).
// ---------------------------------------------------------------------------
#define CONV_CHUNK(CH)                                                         \
    {                                                                          \
        v16h a = *(const v16h*)&Apack[((CH) * 32 + lane) * 16];                \
        const int d = hi ? dtab(2 * (CH) + 1) : dtab(2 * (CH));                \
        v16h b0 = *(const v16h*)&ldsX[row0 + d];                               \
        v16h b1 = *(const v16h*)&ldsX[row1 + d];                               \
        acc0 = __builtin_amdgcn_wmma_f32_16x16x32_f16(                         \
            false, a, false, b0, (short)0, acc0, false, false);                \
        acc1 = __builtin_amdgcn_wmma_f32_16x16x32_f16(                         \
            false, a, false, b1, (short)0, acc1, false, false);                \
    }

__global__ void __launch_bounds__(256) conv3d_wmma_kernel(
    const _Float16* __restrict__ Xin, const _Float16* __restrict__ Apack,
    const float* __restrict__ bias, float* __restrict__ Yout)
{
    __shared__ __align__(32) _Float16 ldsX[LPTS * CC];  // 23040 B
    const int x0 = blockIdx.x * TX;
    const int y0 = blockIdx.y * TY;
    const int z0 = blockIdx.z * TZ;

    // Cooperative halo load (zero-padded SAME boundary)
    for (int p = threadIdx.x; p < LPTS; p += 256) {
        int xi  = p % LXc;
        int rem = p / LXc;
        int yi  = rem % LYc;
        int zi  = rem / LYc;
        int gx = x0 - 1 + xi, gy = y0 - 1 + yi, gz = z0 - 1 + zi;
        v16h val{};
        if ((unsigned)gx < NN && (unsigned)gy < NN && (unsigned)gz < NN)
            val = *(const v16h*)&Xin[(((size_t)gz * NN + gy) * NN + gx) * CC];
        *(v16h*)&ldsX[p * CC] = val;
    }
    __syncthreads();

    const int lane = threadIdx.x & 31;
    const int wave = threadIdx.x >> 5;      // 0..7
    const int lz   = wave >> 2;             // 0..1
    const int ly0  = (wave & 3) * 2;        // 0,2,4,6
    const int n    = lane & 15;             // output x within tile / B column
    const int hi   = lane >> 4;
    const int m0   = hi * 8;                // first oc held by this lane (C/D)

    // Base LDS offsets (halves) for this wave's two output rows.
    const int row0 = ((lz * LYc + ly0) * LXc + n) * CC;
    const int row1 = row0 + LXc * CC;

    v8f acc0, acc1;
    #pragma unroll
    for (int v = 0; v < 8; ++v) acc0[v] = acc1[v] = bias[m0 + v];

    CONV_CHUNK(0)  CONV_CHUNK(1)  CONV_CHUNK(2)  CONV_CHUNK(3)
    CONV_CHUNK(4)  CONV_CHUNK(5)  CONV_CHUNK(6)  CONV_CHUNK(7)
    CONV_CHUNK(8)  CONV_CHUNK(9)  CONV_CHUNK(10) CONV_CHUNK(11)
    CONV_CHUNK(12) CONV_CHUNK(13)

    // D layout: lane holds 8 consecutive oc (m0..m0+7) for spatial point n.
    size_t out0 =
        (((size_t)(z0 + lz) * NN + (y0 + ly0)) * NN + (x0 + n)) * CC + m0;
    *(v8f*)&Yout[out0] = acc0;
    *(v8f*)&Yout[out0 + (size_t)NN * CC] = acc1;
}

// ---------------------------------------------------------------------------
// Deterministic two-stage per-channel instance-norm statistics.
// ---------------------------------------------------------------------------
__global__ void stats_partial_kernel(const float* __restrict__ Y,
                                     float* __restrict__ partial) {
    __shared__ float lds[2][16][16];
    int c = threadIdx.x & 15;
    int g = threadIdx.x >> 4;
    size_t base = (size_t)blockIdx.x * 2048;
    float s = 0.f, ss = 0.f;
    for (int i = 0; i < 128; ++i) {
        float v = Y[(base + (size_t)(g + 16 * i)) * CC + c];
        s += v;
        ss += v * v;
    }
    lds[0][g][c] = s;
    lds[1][g][c] = ss;
    __syncthreads();
    if (threadIdx.x < 16) {
        float S = 0.f, SS = 0.f;
        #pragma unroll
        for (int j = 0; j < 16; ++j) {
            S  += lds[0][j][threadIdx.x];
            SS += lds[1][j][threadIdx.x];
        }
        partial[blockIdx.x * 16 + threadIdx.x] = S;
        partial[1024 * 16 + blockIdx.x * 16 + threadIdx.x] = SS;
    }
}

__global__ void stats_final_kernel(const float* __restrict__ partial,
                                   float* __restrict__ stats) {
    __shared__ float lds[2][16][16];
    int c = threadIdx.x & 15;
    int g = threadIdx.x >> 4;
    float s = 0.f, ss = 0.f;
    for (int i = 0; i < 64; ++i) {
        int b = g + 16 * i;
        s  += partial[b * 16 + c];
        ss += partial[1024 * 16 + b * 16 + c];
    }
    lds[0][g][c] = s;
    lds[1][g][c] = ss;
    __syncthreads();
    if (threadIdx.x < 16) {
        float S = 0.f, SS = 0.f;
        #pragma unroll
        for (int j = 0; j < 16; ++j) {
            S  += lds[0][j][threadIdx.x];
            SS += lds[1][j][threadIdx.x];
        }
        float mean = S / (float)VOX;
        float var  = SS / (float)VOX - mean * mean;
        stats[threadIdx.x]      = mean;
        stats[16 + threadIdx.x] = rsqrtf(var + 1e-5f);
    }
}

// ---------------------------------------------------------------------------
// Fused normalize (+optional residual) + ReLU, f32 -> f16 channel-last.
// ---------------------------------------------------------------------------
__global__ void norm_relu_kernel(const float* __restrict__ Y,
                                 const float* __restrict__ stats,
                                 _Float16* __restrict__ Xdst) {
    size_t t = (size_t)blockIdx.x * blockDim.x + threadIdx.x;
    size_t base = t * 4;
    int c0 = (int)(base & 15);
    v4f y = *(const v4f*)&Y[base];
    v4h o;
    #pragma unroll
    for (int j = 0; j < 4; ++j) {
        float v = (y[j] - stats[c0 + j]) * stats[16 + c0 + j];
        o[j] = (_Float16)(v > 0.f ? v : 0.f);
    }
    *(v4h*)&Xdst[base] = o;
}

__global__ void norm_add_relu_kernel(const float* __restrict__ Y,
                                     const float* __restrict__ stats,
                                     const _Float16* Xres,
                                     _Float16* Xdst) {
    size_t t = (size_t)blockIdx.x * blockDim.x + threadIdx.x;
    size_t base = t * 4;
    int c0 = (int)(base & 15);
    v4f y = *(const v4f*)&Y[base];
    v4h r = *(const v4h*)&Xres[base];
    v4h o;
    #pragma unroll
    for (int j = 0; j < 4; ++j) {
        float v = (y[j] - stats[c0 + j]) * stats[16 + c0 + j] + (float)r[j];
        o[j] = (_Float16)(v > 0.f ? v : 0.f);
    }
    *(v4h*)&Xdst[base] = o;
}

// ---------------------------------------------------------------------------
// Forward projection: angle 0 -> row sums over x; angle pi/2 -> column sums
// over y (exact in f32 given the reference constants).
// out layout: (B=1, C=16, angle=2, N, N)
// ---------------------------------------------------------------------------
__global__ void fwdproject_kernel(const _Float16* __restrict__ X,
                                  float* __restrict__ out) {
    int t = blockIdx.x * blockDim.x + threadIdx.x;  // 2^19 threads
    int c   = t & 15;
    int u   = (t >> 4) & 127;
    int d   = (t >> 11) & 127;
    int ang = t >> 18;
    float acc = 0.f;
    if (ang == 0) {
        size_t base = (((size_t)d * NN + u) * NN) * CC + c;
        for (int x = 0; x < NN; ++x) acc += (float)X[base + (size_t)x * CC];
    } else {
        size_t base = ((size_t)d * NN * NN + u) * CC + c;
        for (int y = 0; y < NN; ++y) acc += (float)X[base + (size_t)y * NN * CC];
    }
    out[((c * 2 + ang) * NN + d) * NN + u] = acc;
}

// ---------------------------------------------------------------------------
extern "C" void kernel_launch(void* const* d_in, const int* in_sizes, int n_in,
                              void* d_out, int out_size, void* d_ws, size_t ws_size,
                              hipStream_t stream) {
    (void)in_sizes; (void)n_in; (void)out_size; (void)ws_size;

    const float* proj = (const float*)d_in[0];
    const float* w[4]  = {(const float*)d_in[1], (const float*)d_in[3],
                          (const float*)d_in[5], (const float*)d_in[7]};
    const float* bs[4] = {(const float*)d_in[2], (const float*)d_in[4],
                          (const float*)d_in[6], (const float*)d_in[8]};

    char* ws = (char*)d_ws;
    _Float16* X0    = (_Float16*)ws;                              // 64 MB
    _Float16* X1    = (_Float16*)(ws + (size_t)67108864);         // 64 MB
    float*    Y     = (float*)   (ws + (size_t)134217728);        // 128 MB
    _Float16* Apack = (_Float16*)(ws + (size_t)268435456);        // 56 KB
    float*    part  = (float*)   (ws + (size_t)268492800);        // 128 KB
    float*    stats = (float*)   (ws + (size_t)268623872);        // 128 B

    for (int i = 0; i < 4; ++i)
        pack_weights_kernel<<<28, 256, 0, stream>>>(
            w[i], Apack + (size_t)i * NCHUNK * 32 * 16);

    backproject_kernel<<<VOX / 256, 256, 0, stream>>>(proj, X0);

    dim3 cgrid(NN / TX, NN / TY, NN / TZ);  // (8, 16, 64)
    for (int blk = 0; blk < 2; ++blk) {
        int c1 = blk * 2, c2 = blk * 2 + 1;
        conv3d_wmma_kernel<<<cgrid, 256, 0, stream>>>(
            X0, Apack + (size_t)c1 * NCHUNK * 32 * 16, bs[c1], Y);
        stats_partial_kernel<<<1024, 256, 0, stream>>>(Y, part);
        stats_final_kernel<<<1, 256, 0, stream>>>(part, stats);
        norm_relu_kernel<<<32768, 256, 0, stream>>>(Y, stats, X1);

        conv3d_wmma_kernel<<<cgrid, 256, 0, stream>>>(
            X1, Apack + (size_t)c2 * NCHUNK * 32 * 16, bs[c2], Y);
        stats_partial_kernel<<<1024, 256, 0, stream>>>(Y, part);
        stats_final_kernel<<<1, 256, 0, stream>>>(part, stats);
        norm_add_relu_kernel<<<32768, 256, 0, stream>>>(Y, stats, X0, X0);
    }

    fwdproject_kernel<<<2048, 256, 0, stream>>>(X0, (float*)d_out);
}
